// GATv2_21543555957026
// MI455X (gfx1250) — matrix operand order repaired and checked by reading, not compiled, not run on hardware
//
#include <hip/hip_runtime.h>
#include <hip/hip_bf16.h>

// GATv2 fused kernel for gfx1250 (MI455X).
// N=4096 nodes, NNI=64 neighbors, NIN=512, NOUT=512, LeakyReLU slope 0.2.
//
// Kernel 1: repack W_w (f32, [512][512]) -> bf16 WMMA B-fragments in d_ws.
// Kernel 2: per node n:
//   - stage hjs[n] (64x512 f32, non-temporal stream) + hi[n] (row 64) into LDS
//     as bf16 (rows 65..79 zero)
//   - GEMM M=80 x N=512 x K=512 via v_wmma_f32_16x16x32_bf16 (f32 acc), + bias
//   - e_j = sum_o a_w[o]*lrelu(Whi[o]+Whj[j][o]); softmax over 64 j
//   - agg[o] = sum_j alpha_j*Whj[j][o]; sigmoid -> out[n][o] (non-temporal)

#define NNODE 4096
#define NNI   64
#define NIN   512
#define NOUT  512
#define PITCH 520                 // bf16 row pitch in LDS (padded, 16B-aligned rows)
#define NSLOPE 0.2f

typedef __bf16        v16bf __attribute__((ext_vector_type(16)));
typedef float         v8f   __attribute__((ext_vector_type(8)));
typedef unsigned int  v8u   __attribute__((ext_vector_type(8)));
typedef float         v4f   __attribute__((ext_vector_type(4)));

__device__ __forceinline__ unsigned bf16pk(float a, float b) {
    unsigned ua = __float_as_uint(a), ub = __float_as_uint(b);
    ua += 0x7FFFu + ((ua >> 16) & 1u);          // round-to-nearest-even
    ub += 0x7FFFu + ((ub >> 16) & 1u);
    return (ua >> 16) | (ub & 0xFFFF0000u);
}

__device__ __forceinline__ v16bf make_frag(uint4 lo, uint4 hi) {
    v8u r;
    r[0] = lo.x; r[1] = lo.y; r[2] = lo.z; r[3] = lo.w;
    r[4] = hi.x; r[5] = hi.y; r[6] = hi.z; r[7] = hi.w;
    return __builtin_bit_cast(v16bf, r);
}

// ---------------- Kernel 1: W -> bf16 B-fragment layout ----------------
// Fragment (ntile, kc): lane l holds N = ntile*16 + l%16; dword d holds K pair
// (k0, k0+1), k0 = kc*32 + (l/16)*8 + (d/4)*16 + (d%4)*2.  B[k][n] = W[n][k]  (Whj = hjs @ W^T).
// Flat dword index: (((ntile*16 + kc)*32 + lane)*8 + d)
__global__ void __launch_bounds__(256) gat_prep_wfrag(const float* __restrict__ W,
                                                      unsigned* __restrict__ wf) {
    int t = blockIdx.x * 256 + threadIdx.x;       // 131072 dwords total
    int d     =  t        & 7;
    int lane  = (t >> 3)  & 31;
    int kc    = (t >> 8)  & 15;
    int ntile =  t >> 12;
    int k0 = kc * 32 + (lane >> 4) * 8 + (d >> 2) * 16 + (d & 3) * 2;
    int n  = ntile * 16 + (lane & 15);
    wf[t] = bf16pk(W[(size_t)n * NIN + k0], W[(size_t)n * NIN + k0 + 1]);
}

// ---------------- Kernel 2: fused GATv2 ----------------
__global__ void __launch_bounds__(256) gat_fused(const float* __restrict__ hi,
                                                 const float* __restrict__ hjs,
                                                 const uint4* __restrict__ wfrag,
                                                 const float* __restrict__ W_b,
                                                 const float* __restrict__ a_w,
                                                 float* __restrict__ out) {
    extern __shared__ char smem[];
    unsigned short* Asm    = (unsigned short*)smem;               // [80][PITCH] bf16
    float*          whiLds = (float*)(smem + 80 * PITCH * 2);     // [512]
    float*          eLds   = (float*)(smem + 80 * PITCH * 2 + NOUT * 4); // [64]

    const int tid  = threadIdx.x;
    const int lane = tid & 31;
    const int w    = tid >> 5;            // wave id 0..7
    const int wn0  = w * 64;              // this wave's N base (4 n-tiles)
    const int node = blockIdx.x;
    const int nlo  = lane & 15;
    const int half = lane >> 4;

    // ---- Stage hjs[node] -> bf16 LDS (rows 0..63), hi[node] -> row 64, zeros 65..79
    // hjs is a 512 MB single-use stream: use non-temporal loads so the L2-resident
    // weight fragments / bias / a_w are not evicted.
    {
        const v4f* src = (const v4f*)(hjs + (size_t)node * NNI * NIN); // 8192 f4
        #pragma unroll 4
        for (int idx = tid; idx < NNI * (NIN / 4); idx += 256) {
            int m  = idx >> 7;            // 128 float4 per row
            int kq = idx & 127;
            v4f v = __builtin_nontemporal_load(src + idx);
            uint2 p; p.x = bf16pk(v.x, v.y); p.y = bf16pk(v.z, v.w);
            *(uint2*)&Asm[m * PITCH + kq * 4] = p;
        }
        const v4f* hs = (const v4f*)(hi + (size_t)node * NIN);
        if (tid < 128) {
            v4f v = __builtin_nontemporal_load(hs + tid);
            uint2 p; p.x = bf16pk(v.x, v.y); p.y = bf16pk(v.z, v.w);
            *(uint2*)&Asm[64 * PITCH + tid * 4] = p;
        }
        for (int idx = tid; idx < 15 * 128; idx += 256) {
            int m = 65 + (idx >> 7), kq = idx & 127;
            *(uint2*)&Asm[m * PITCH + kq * 4] = make_uint2(0u, 0u);
        }
        if (tid < NNI) eLds[tid] = 0.0f;
    }
    __syncthreads();

    // ---- GEMM: 5 M-tiles x 4 N-tiles per wave, K = 16 chunks of 32
    v8f acc[5][4];
    #pragma unroll
    for (int mt = 0; mt < 5; ++mt)
        #pragma unroll
        for (int nt = 0; nt < 4; ++nt)
            acc[mt][nt] = (v8f)0.0f;

    for (int kc = 0; kc < 16; ++kc) {
        v16bf Bf[4];
        #pragma unroll
        for (int nt = 0; nt < 4; ++nt) {
            const uint4* p = wfrag + (size_t)((((w * 4 + nt) * 16 + kc) * 32 + lane) * 2);
            Bf[nt] = make_frag(p[0], p[1]);
        }
        const int kstart = kc * 32 + half * 8;
        #pragma unroll
        for (int mt = 0; mt < 5; ++mt) {
            const unsigned short* ap = Asm + (mt * 16 + nlo) * PITCH + kstart;
            v16bf Af = make_frag(*(const uint4*)ap, *(const uint4*)(ap + 16));
            #pragma unroll
            for (int nt = 0; nt < 4; ++nt) {
                acc[mt][nt] = __builtin_amdgcn_wmma_f32_16x16x32_bf16(
                    false, Af, false, Bf[nt], (short)0, acc[mt][nt], false, false);
            }
        }
    }

    // ---- bias + publish Whi row (M=64 lives in mt=4, vgpr 0, lanes 0..15)
    float aw4[4], bias4[4];
    #pragma unroll
    for (int nt = 0; nt < 4; ++nt) {
        int n = wn0 + nt * 16 + nlo;
        bias4[nt] = W_b[n];
        aw4[nt]   = a_w[n];
        #pragma unroll
        for (int mt = 0; mt < 5; ++mt)
            #pragma unroll
            for (int i = 0; i < 8; ++i)
                acc[mt][nt][i] += bias4[nt];
        if (lane < 16) whiLds[n] = acc[4][nt][0];
    }
    __syncthreads();

    // ---- edge scores: e[j] = sum_o a_w[o] * lrelu(Whi[o] + Whj[j][o])
    float whi4[4];
    #pragma unroll
    for (int nt = 0; nt < 4; ++nt) whi4[nt] = whiLds[wn0 + nt * 16 + nlo];

    #pragma unroll
    for (int mt = 0; mt < 4; ++mt) {
        #pragma unroll
        for (int i = 0; i < 8; ++i) {
            int m = mt * 16 + i + half * 8;
            float part = 0.0f;
            #pragma unroll
            for (int nt = 0; nt < 4; ++nt) {
                float pre = whi4[nt] + acc[mt][nt][i];
                float act = (pre >= 0.0f) ? pre : NSLOPE * pre;
                part += aw4[nt] * act;
            }
            part += __shfl_xor(part, 1);
            part += __shfl_xor(part, 2);
            part += __shfl_xor(part, 4);
            part += __shfl_xor(part, 8);
            if (nlo == 0) atomicAdd(&eLds[m], part);   // ds_add_f32, cross-wave
        }
    }
    __syncthreads();

    // ---- softmax over the 64 neighbors (wave 0 only)
    if (tid < 32) {
        float e0 = eLds[tid], e1 = eLds[tid + 32];
        float mx = fmaxf(e0, e1);
        #pragma unroll
        for (int off = 1; off < 32; off <<= 1) mx = fmaxf(mx, __shfl_xor(mx, off));
        float x0 = __expf(e0 - mx), x1 = __expf(e1 - mx);
        float s = x0 + x1;
        #pragma unroll
        for (int off = 1; off < 32; off <<= 1) s += __shfl_xor(s, off);
        float inv = 1.0f / s;
        eLds[tid] = x0 * inv; eLds[tid + 32] = x1 * inv;   // alpha in place
    }
    __syncthreads();

    // ---- aggregation + sigmoid: agg[n] = sum_j alpha[j]*Whj[j][n]
    #pragma unroll
    for (int nt = 0; nt < 4; ++nt) {
        float agg = 0.0f;
        #pragma unroll
        for (int mt = 0; mt < 4; ++mt)
            #pragma unroll
            for (int i = 0; i < 8; ++i)
                agg += eLds[mt * 16 + i + half * 8] * acc[mt][nt][i];
        agg += __shfl_xor(agg, 16);                        // merge M halves
        float r = 1.0f / (1.0f + __expf(-agg));
        if (lane < 16)
            __builtin_nontemporal_store(r, out + (size_t)node * NOUT + wn0 + nt * 16 + lane);
    }
}

extern "C" void kernel_launch(void* const* d_in, const int* in_sizes, int n_in,
                              void* d_out, int out_size, void* d_ws, size_t ws_size,
                              hipStream_t stream) {
    const float* hi  = (const float*)d_in[0];
    const float* hjs = (const float*)d_in[1];
    const float* W_w = (const float*)d_in[2];
    const float* W_b = (const float*)d_in[3];
    const float* a_w = (const float*)d_in[4];
    float* out = (float*)d_out;
    unsigned* wf = (unsigned*)d_ws;                        // 512 KB bf16 W fragments

    gat_prep_wfrag<<<(NOUT * NIN / 2) / 256, 256, 0, stream>>>(W_w, wf);

    const size_t lds = 80 * PITCH * 2 + NOUT * 4 + NNI * 4;  // ~85.5 KB
    gat_fused<<<NNODE, 256, lds, stream>>>(hi, hjs, (const uint4*)wf, W_b, a_w, out);
}
